// MultHeadAttention_12627203851088
// MI455X (gfx1250) — compile-verified, matrix operand order checked by reading, not run on hardware
//
#include <hip/hip_runtime.h>
#include <hip/hip_bf16.h>

#define B_ 4
#define S_ 2048
#define E_ 1024
#define H_ 16
#define D_ 64

typedef __attribute__((ext_vector_type(16))) _Float16 v16h;
typedef __attribute__((ext_vector_type(8)))  float    v8f;

union FragU {
    v16h     v;
    _Float16 h[16];
    uint4    q[2];
};

__device__ inline v8f wmma_f16(v16h a, v16h b, v8f c) {
    // D(f32 16x16) = A(f16 16x32) * B(f16 32x16) + C
    return __builtin_amdgcn_wmma_f32_16x16x32_f16(
        /*neg_a=*/false, a, /*neg_b=*/false, b,
        /*c_mod=*/(short)0, c, /*reuse_a=*/false, /*reuse_b=*/false);
}

// ---------------------------------------------------------------------------
// Tiled GEMM:  Out[M=B*S, N=E] = A[M, E] @ W[E, N] + bias[N]
// A_F32:    A is f32 (projection inputs) vs f16 (attention output)
// HEAD_OUT: write f16 into [B,H,S,D] head layout vs f32 row-major [M,N]
// Block: 256 threads (8 waves). Block tile 128x128, wave tile 64x32,
// K step 64 -> 16 WMMAs per barrier pair.
// ---------------------------------------------------------------------------
template <bool A_F32, bool HEAD_OUT>
__global__ void __launch_bounds__(256)
gemm_kernel(const void* __restrict__ Aptr, const float* __restrict__ W,
            const float* __restrict__ bias, void* __restrict__ Out)
{
    __shared__ __align__(16) _Float16 As[128][64];   // [m][k]
    __shared__ __align__(16) _Float16 Bs[128][64];   // [n][k] (transposed store)

    const int tid   = threadIdx.x;
    const int lane  = tid & 31;
    const int wave  = tid >> 5;       // 0..7
    const int waveM = wave >> 2;      // 0..1  -> 64-row slab
    const int waveN = wave & 3;       // 0..3  -> 32-col slab
    const int l16   = lane & 15;
    const int hsel  = lane >> 4;      // which 16-lane half

    const int blockM = blockIdx.y * 128;
    const int blockN = blockIdx.x * 128;

    v8f zero = {};
    v8f acc[4][2];
#pragma unroll
    for (int mt = 0; mt < 4; ++mt)
#pragma unroll
        for (int nt = 0; nt < 2; ++nt) acc[mt][nt] = zero;

    for (int k0 = 0; k0 < E_; k0 += 64) {
        // ---- cooperative load of A tile (128x64) into LDS as f16 ----
        {
            const int row = tid >> 1;          // 0..127
            const int cg  = (tid & 1) * 32;    // 0 or 32
            const int gr  = blockM + row;
            if (A_F32) {
                const float* A = (const float*)Aptr;
                const float4* src = (const float4*)(A + (size_t)gr * E_ + k0 + cg);
                _Float16 tmp[32];
#pragma unroll
                for (int i = 0; i < 8; ++i) {
                    float4 f = src[i];
                    tmp[i * 4 + 0] = (_Float16)f.x;
                    tmp[i * 4 + 1] = (_Float16)f.y;
                    tmp[i * 4 + 2] = (_Float16)f.z;
                    tmp[i * 4 + 3] = (_Float16)f.w;
                }
                uint4* dst = (uint4*)&As[row][cg];
#pragma unroll
                for (int i = 0; i < 4; ++i) dst[i] = ((const uint4*)tmp)[i];
                if (k0 + 64 < E_)
                    __builtin_prefetch(A + (size_t)gr * E_ + k0 + 64 + cg, 0, 1);
            } else {
                const _Float16* A = (const _Float16*)Aptr;
                const uint4* src = (const uint4*)(A + (size_t)gr * E_ + k0 + cg);
                uint4* dst = (uint4*)&As[row][cg];
#pragma unroll
                for (int i = 0; i < 4; ++i) dst[i] = src[i];
                if (k0 + 64 < E_)
                    __builtin_prefetch(A + (size_t)gr * E_ + k0 + 64 + cg, 0, 1);
            }
        }
        // ---- cooperative load of W tile (64x128), transpose into Bs[n][k] ----
        {
            const int krow = tid >> 2;           // 0..63
            const int cg   = (tid & 3) * 32;     // 0..96
            const float4* src = (const float4*)(W + (size_t)(k0 + krow) * E_ + blockN + cg);
#pragma unroll
            for (int i = 0; i < 8; ++i) {
                float4 f = src[i];
                Bs[cg + i * 4 + 0][krow] = (_Float16)f.x;
                Bs[cg + i * 4 + 1][krow] = (_Float16)f.y;
                Bs[cg + i * 4 + 2][krow] = (_Float16)f.z;
                Bs[cg + i * 4 + 3][krow] = (_Float16)f.w;
            }
            if (k0 + 64 < E_)
                __builtin_prefetch(W + (size_t)(k0 + 64 + krow) * E_ + blockN + cg, 0, 1);
        }
        __syncthreads();

#pragma unroll
        for (int kk = 0; kk < 64; kk += 32) {
            // ---- B fragments (K=32 striped per lane-half) ----
            FragU bfrag[2];
#pragma unroll
            for (int nt = 0; nt < 2; ++nt) {
                const int n = waveN * 32 + nt * 16 + l16;
                bfrag[nt].q[0] = *(const uint4*)&Bs[n][kk + hsel * 8];
                bfrag[nt].q[1] = *(const uint4*)&Bs[n][kk + 16 + hsel * 8];
            }
            // ---- A fragments + WMMA ----
#pragma unroll
            for (int mt = 0; mt < 4; ++mt) {
                FragU afrag;
                const int m = waveM * 64 + mt * 16 + l16;
                afrag.q[0] = *(const uint4*)&As[m][kk + hsel * 8];
                afrag.q[1] = *(const uint4*)&As[m][kk + 16 + hsel * 8];
#pragma unroll
                for (int nt = 0; nt < 2; ++nt)
                    acc[mt][nt] = wmma_f16(afrag.v, bfrag[nt].v, acc[mt][nt]);
            }
        }
        __syncthreads();
    }

    // ---- epilogue: bias add + store (C layout: comp r -> M=r / r+8 per lane half) ----
#pragma unroll
    for (int mt = 0; mt < 4; ++mt) {
#pragma unroll
        for (int nt = 0; nt < 2; ++nt) {
#pragma unroll
            for (int r = 0; r < 8; ++r) {
                const int m = blockM + waveM * 64 + mt * 16 + r + hsel * 8;
                const int n = blockN + waveN * 32 + nt * 16 + l16;
                const float val = acc[mt][nt][r] + bias[n];
                if (HEAD_OUT) {
                    const int b = m / S_, s = m % S_;
                    const int h = n / D_, d = n % D_;
                    ((_Float16*)Out)[(((size_t)(b * H_ + h)) * S_ + s) * D_ + d] = (_Float16)val;
                } else {
                    ((float*)Out)[(size_t)m * E_ + n] = val;
                }
            }
        }
    }
}

// ---------------------------------------------------------------------------
// Causal flash attention: Qh/Kh/Vh are f16 [B,H,S,D]; output f16 [B,S,E].
// Block = one (b,h) x 128 query rows; wave w owns rows [base+16w, base+16w+15].
// Key loop in chunks of 32; K staged [key][d], V staged TRANSPOSED [d][key]
// so every WMMA fragment load is a pair of ds_load_b128s.
// ---------------------------------------------------------------------------
__global__ void __launch_bounds__(256)
attn_kernel(const _Float16* __restrict__ Qh, const _Float16* __restrict__ Kh,
            const _Float16* __restrict__ Vh, _Float16* __restrict__ Oh)
{
    __shared__ __align__(16) _Float16 Ks[32][64];      // [key][d]
    __shared__ __align__(16) _Float16 VsT[64][32];     // [d][key]  (transposed)
    __shared__ __align__(16) _Float16 Ps[8][16][32];   // per-wave P tile

    const int tid  = threadIdx.x;
    const int lane = tid & 31;
    const int wave = tid >> 5;
    const int l16  = lane & 15;
    const int hsel = lane >> 4;

    const int bh = blockIdx.y;           // 0..B*H-1
    const int b  = bh / H_;
    const int h  = bh % H_;
    const int rowblk  = blockIdx.x;      // 0..S/128-1
    const int rowbase = rowblk * 128 + wave * 16;

    const _Float16* Qbase = Qh + (size_t)bh * S_ * D_;
    const _Float16* Kbase = Kh + (size_t)bh * S_ * D_;
    const _Float16* Vbase = Vh + (size_t)bh * S_ * D_;

    // Q fragments: two K=32 slices of the 16x64 Q row tile (direct from global)
    FragU qf[2];
#pragma unroll
    for (int dc = 0; dc < 2; ++dc) {
        const int row = rowbase + l16;
        qf[dc].q[0] = *(const uint4*)(Qbase + (size_t)row * D_ + dc * 32 + hsel * 8);
        qf[dc].q[1] = *(const uint4*)(Qbase + (size_t)row * D_ + dc * 32 + 16 + hsel * 8);
    }

    v8f zero = {};
    v8f oacc[4];
#pragma unroll
    for (int nt = 0; nt < 4; ++nt) oacc[nt] = zero;

    float mrow[8], lrow[8];
#pragma unroll
    for (int r = 0; r < 8; ++r) { mrow[r] = -1e30f; lrow[r] = 0.0f; }

    const float scale = 0.125f;                 // 1/sqrt(64)
    const int tmax = rowblk * 128 + 96;         // last chunk start (inclusive)

    for (int t = 0; t <= tmax; t += 32) {
        // cooperative stage: K as [key][d] (uint4 store), V transposed to [d][key]
        {
            const int idx = tid * 8;
            const int kr = idx >> 6;            // key row 0..31
            const int kc = idx & 63;            // d col  0,8,..,56
            *(uint4*)&Ks[kr][kc] = *(const uint4*)(Kbase + (size_t)(t + kr) * D_ + kc);
            FragU vt;
            vt.q[0] = *(const uint4*)(Vbase + (size_t)(t + kr) * D_ + kc);
#pragma unroll
            for (int j = 0; j < 8; ++j)
                VsT[kc + j][kr] = vt.h[j];
            if (t + 32 <= tmax) {
                __builtin_prefetch(Kbase + (size_t)(t + 32 + kr) * D_ + kc, 0, 1);
                __builtin_prefetch(Vbase + (size_t)(t + 32 + kr) * D_ + kc, 0, 1);
            }
        }
        __syncthreads();

        if (t <= rowbase + 15) {   // chunk not entirely masked for this wave
            // ---- scores: S[16x32] = Q[16x64] @ K^T[64x32], 4 WMMAs ----
            v8f sacc[2] = { zero, zero };
#pragma unroll
            for (int ct = 0; ct < 2; ++ct) {
#pragma unroll
                for (int dc = 0; dc < 2; ++dc) {
                    FragU kf;   // B-matrix: rows = d, cols = key
                    const int key = ct * 16 + l16;
                    kf.q[0] = *(const uint4*)&Ks[key][dc * 32 + hsel * 8];
                    kf.q[1] = *(const uint4*)&Ks[key][dc * 32 + 16 + hsel * 8];
                    sacc[ct] = wmma_f16(qf[dc].v, kf.v, sacc[ct]);
                }
            }
            // ---- online softmax (row stats uniform across each 16-lane half) ----
#pragma unroll
            for (int r = 0; r < 8; ++r) {
                const int row = rowbase + r + hsel * 8;
                const int c0 = t + l16;
                const int c1 = t + 16 + l16;
                float x0 = (c0 <= row) ? sacc[0][r] * scale : -1e30f;
                float x1 = (c1 <= row) ? sacc[1][r] * scale : -1e30f;
                float mx = fmaxf(x0, x1);
#pragma unroll
                for (int msk = 1; msk < 16; msk <<= 1)
                    mx = fmaxf(mx, __shfl_xor(mx, msk, 32));
                const float mnew = fmaxf(mrow[r], mx);
                const float sf = __expf(mrow[r] - mnew);
                const float p0 = __expf(x0 - mnew);
                const float p1 = __expf(x1 - mnew);
                float rs = p0 + p1;
#pragma unroll
                for (int msk = 1; msk < 16; msk <<= 1)
                    rs += __shfl_xor(rs, msk, 32);
                lrow[r] = lrow[r] * sf + rs;
                mrow[r] = mnew;
#pragma unroll
                for (int nt = 0; nt < 4; ++nt) oacc[nt][r] *= sf;
                const int prow = r + hsel * 8;
                Ps[wave][prow][l16]      = (_Float16)p0;
                Ps[wave][prow][16 + l16] = (_Float16)p1;
            }
            // ---- O += P[16x32] @ V[32x64], 4 WMMAs (DS ops in-order per wave) ----
            FragU pf;
            pf.q[0] = *(const uint4*)&Ps[wave][l16][hsel * 8];
            pf.q[1] = *(const uint4*)&Ps[wave][l16][16 + hsel * 8];
#pragma unroll
            for (int nt = 0; nt < 4; ++nt) {
                FragU vf;   // B-matrix: rows = key, cols = d (contiguous via VsT)
                const int d = nt * 16 + l16;
                vf.q[0] = *(const uint4*)&VsT[d][hsel * 8];
                vf.q[1] = *(const uint4*)&VsT[d][16 + hsel * 8];
                oacc[nt] = wmma_f16(pf.v, vf.v, oacc[nt]);
            }
        }
        __syncthreads();
    }

    // ---- normalize + store to [B,S,E] f16 ----
#pragma unroll
    for (int r = 0; r < 8; ++r) {
        const float inv = (lrow[r] > 0.0f) ? 1.0f / lrow[r] : 0.0f;
        const int srow = rowbase + r + hsel * 8;
#pragma unroll
        for (int nt = 0; nt < 4; ++nt) {
            const int d = nt * 16 + l16;
            Oh[((size_t)b * S_ + srow) * E_ + h * D_ + d] = (_Float16)(oacc[nt][r] * inv);
        }
    }
}

extern "C" void kernel_launch(void* const* d_in, const int* in_sizes, int n_in,
                              void* d_out, int out_size, void* d_ws, size_t ws_size,
                              hipStream_t stream) {
    (void)in_sizes; (void)n_in; (void)out_size; (void)ws_size;
    const float* q   = (const float*)d_in[0];
    const float* k   = (const float*)d_in[1];
    const float* v   = (const float*)d_in[2];
    const float* w_q = (const float*)d_in[3];
    const float* b_q = (const float*)d_in[4];
    const float* w_k = (const float*)d_in[5];
    const float* b_k = (const float*)d_in[6];
    const float* w_v = (const float*)d_in[7];
    const float* b_v = (const float*)d_in[8];
    const float* w_o = (const float*)d_in[9];
    const float* b_o = (const float*)d_in[10];

    const size_t elems = (size_t)B_ * S_ * E_;   // 8Mi elements
    _Float16* Qh = (_Float16*)d_ws;
    _Float16* Kh = Qh + elems;
    _Float16* Vh = Kh + elems;
    _Float16* AO = Vh + elems;                   // attention output, [B,S,E] f16

    const dim3 gblk(256);
    const dim3 ggrid(E_ / 128, (B_ * S_) / 128);          // (8, 64)

    gemm_kernel<true,  true ><<<ggrid, gblk, 0, stream>>>(q, w_q, b_q, Qh);
    gemm_kernel<true,  true ><<<ggrid, gblk, 0, stream>>>(k, w_k, b_k, Kh);
    gemm_kernel<true,  true ><<<ggrid, gblk, 0, stream>>>(v, w_v, b_v, Vh);

    attn_kernel<<<dim3(S_ / 128, B_ * H_), gblk, 0, stream>>>(Qh, Kh, Vh, AO);

    gemm_kernel<false, false><<<ggrid, gblk, 0, stream>>>(AO, w_o, b_o, (float*)d_out);
}
